// ConceptNAMNary_44650480010036
// MI455X (gfx1250) — matrix operand order repaired and checked by reading, not compiled, not run on hardware
//
#include <hip/hip_runtime.h>
#include <hip/hip_bf16.h>

typedef __attribute__((ext_vector_type(16))) _Float16 v16h;
typedef __attribute__((ext_vector_type(8)))  float    v8f;
typedef __attribute__((ext_vector_type(2)))  _Float16 h2v;

#define BB   2048
#define NCC  32
#define MM   496
#define H1C  64
#define H2C  32
#define CLSC 10
#define EPSF 1e-5f
#define BTILE 256   // batch rows per block (8 waves * 32 rows)

union HFrag { v16h v; _Float16 h[16]; unsigned u[8]; };

// xor-butterfly add via ds_swizzle (group-of-32 mode: offset = xor<<10 | and_mask 0x1f)
#define SWIZ_ADD(val, xm)                                                         \
    (val) += __int_as_float(                                                      \
        __builtin_amdgcn_ds_swizzle(__float_as_int(val), ((xm) << 10) | 0x1f))

// cvt two f32 to packed f16 then packed ReLU (v_cvt_pk_f16_f32 + v_pk_max_num_f16)
__device__ inline unsigned relu_pk_f16(float a, float b) {
    h2v p; p.x = (_Float16)a; p.y = (_Float16)b;
    h2v z = (_Float16)0.0f;
    p = __builtin_elementwise_max(p, z);
    union { h2v v; unsigned u; } c; c.v = p;
    return c.u;
}

// Fused per-pair MLP: layer1 (VALU) -> layer2 (WMMA f16->f32) -> layer3 + lane reduce.
// One block = one m (pair) x 256 batch rows. Output: o_ws[m*BB + b] scalar field.
__global__ __launch_bounds__(256) void nam_pair_kernel(
    const float* __restrict__ x,  const int*   __restrict__ pair_idx,
    const float* __restrict__ W1, const float* __restrict__ b1,
    const float* __restrict__ g1, const float* __restrict__ be1,
    const float* __restrict__ m1, const float* __restrict__ v1,
    const float* __restrict__ W2, const float* __restrict__ b2,
    const float* __restrict__ g2, const float* __restrict__ be2,
    const float* __restrict__ m2, const float* __restrict__ v2,
    const float* __restrict__ W3, const float* __restrict__ b3,
    float* __restrict__ o_ws)
{
    __shared__ float P0[H1C], P1[H1C], Q[H1C];     // folded layer1+BN1 constants
    __shared__ float A2s[H2C], C2s[H2C], W3s[H2C]; // folded BN2 + W3
    __shared__ _Float16 w2h[H2C * H1C];            // W2[m] in f16, natural (n,k) order

    const int m     = blockIdx.x;   // 0..495
    const int btile = blockIdx.y;   // 0..7
    const int t     = threadIdx.x;

    // ---- fold parameters into LDS ----
    if (t < H1C) {
        const int k = t;
        const float rs = rsqrtf(v1[m*H1C + k] + EPSF);
        const float a  = g1[m*H1C + k] * rs;
        P0[k] = a * W1[(m*H1C + k)*2 + 0];
        P1[k] = a * W1[(m*H1C + k)*2 + 1];
        Q[k]  = (b1[m*H1C + k] - m1[m*H1C + k]) * a + be1[m*H1C + k];
    } else if (t < H1C + H2C) {
        const int n = t - H1C;
        const float rs = rsqrtf(v2[m*H2C + n] + EPSF);
        const float a  = g2[m*H2C + n] * rs;
        A2s[n] = a;
        C2s[n] = (b2[m*H2C + n] - m2[m*H2C + n]) * a + be2[m*H2C + n];
        W3s[n] = W3[m*H2C + n];
    }
    // convert W2[m] (32x64 fp32, 2048 elems) to f16 in LDS; 8 per thread
    #pragma unroll
    for (int i = 0; i < 8; ++i) {
        const int idx = t + 256*i;
        w2h[idx] = (_Float16)W2[m*(H2C*H1C) + idx];
    }
    __syncthreads();

    const int wave = t >> 5;       // 0..7
    const int lane = t & 31;
    const int lg   = lane & 15;    // index within 16-lane group
    const int lh   = lane >> 4;    // which half of the wave

    const int rbase = btile*BTILE + wave*32;   // 32 rows per wave (2 row-tiles of 16)
    const int i0 = pair_idx[m*2 + 0];
    const int i1 = pair_idx[m*2 + 1];

    // ---- B fragments (32x16 f16): lane = column n, halves = K pairs 2v + 16*lh ----
    // Shared by both row-tiles: loaded once per wave.
    HFrag b00, b10, b01, b11;   // b[kblock][ntile]
    const unsigned* w2u = (const unsigned*)w2h;
    const int n0 = lg, n1 = lg + 16;
    #pragma unroll
    for (int v = 0; v < 8; ++v) {
        const int kb = 2*v + 16*lh;                  // K within a 32-wide block (even)
        b00.u[v] = w2u[(n0*H1C + kb)      >> 1];
        b10.u[v] = w2u[(n0*H1C + kb + 32) >> 1];
        b01.u[v] = w2u[(n1*H1C + kb)      >> 1];
        b11.u[v] = w2u[(n1*H1C + kb + 32) >> 1];
    }

    const float A2_0 = A2s[n0], C2_0 = C2s[n0], W3_0 = W3s[n0];
    const float A2_1 = A2s[n1], C2_1 = C2s[n1], W3_1 = W3s[n1];
    const float b3m  = b3[m];
    float* orow_ws = o_ws + m*BB;    // [m][B] layout: near-coalesced block stores

    #pragma unroll
    for (int rt = 0; rt < 2; ++rt) {
        const int rowbase = rbase + 16*rt;
        const int row = rowbase + lg;  // A-matrix row (both lane halves cover same rows)
        const float x0 = x[row*NCC + i0];
        const float x1 = x[row*NCC + i1];

        // ---- A fragments: layer1 + BN1 + packed-f16 ReLU, in WMMA layout ----
        // half j of 16x32 f16 A frag maps to K = j + 16*lh; pairs are contiguous K.
        HFrag a0, a1;
        #pragma unroll
        for (int j = 0; j < 16; j += 2) {
            const int k = j + 16*lh;
            float u0 = fmaf(x0, P0[k],     fmaf(x1, P1[k],     Q[k]));
            float u1 = fmaf(x0, P0[k + 1], fmaf(x1, P1[k + 1], Q[k + 1]));
            a0.u[j >> 1] = relu_pk_f16(u0, u1);
            const int k2 = k + 32;
            float w0 = fmaf(x0, P0[k2],     fmaf(x1, P1[k2],     Q[k2]));
            float w1 = fmaf(x0, P0[k2 + 1], fmaf(x1, P1[k2 + 1], Q[k2 + 1]));
            a1.u[j >> 1] = relu_pk_f16(w0, w1);
        }

        // ---- layer2: (16x64)x(64x32) via 4 chained v_wmma_f32_16x16x32_f16 ----
        v8f d0 = {}; v8f d1 = {};
        d0 = __builtin_amdgcn_wmma_f32_16x16x32_f16(false, a0.v, false, b00.v, (short)0, d0, false, false);
        d1 = __builtin_amdgcn_wmma_f32_16x16x32_f16(false, a0.v, false, b01.v, (short)0, d1, false, false);
        d0 = __builtin_amdgcn_wmma_f32_16x16x32_f16(false, a1.v, false, b10.v, (short)0, d0, false, false);
        d1 = __builtin_amdgcn_wmma_f32_16x16x32_f16(false, a1.v, false, b11.v, (short)0, d1, false, false);

        // ---- BN2 + ReLU + layer3 dot ----
        float val[8];
        #pragma unroll
        for (int r = 0; r < 8; ++r) {
            float h0 = fmaf(d0[r], A2_0, C2_0); h0 = h0 > 0.f ? h0 : 0.f;
            float h1 = fmaf(d1[r], A2_1, C2_1); h1 = h1 > 0.f ? h1 : 0.f;
            val[r] = fmaf(h0, W3_0, h1 * W3_1);
        }

        // ---- split-butterfly reduce over the 16-lane group (15 swizzles) ----
        #pragma unroll
        for (int r = 0; r < 8; ++r) SWIZ_ADD(val[r], 8);
        float c4[4];
        #pragma unroll
        for (int i = 0; i < 4; ++i) c4[i] = (lg & 8) ? val[i + 4] : val[i];
        #pragma unroll
        for (int i = 0; i < 4; ++i) SWIZ_ADD(c4[i], 4);
        float c2[2];
        c2[0] = (lg & 4) ? c4[2] : c4[0];
        c2[1] = (lg & 4) ? c4[3] : c4[1];
        SWIZ_ADD(c2[0], 2);
        SWIZ_ADD(c2[1], 2);
        float e = (lg & 2) ? c2[1] : c2[0];
        SWIZ_ADD(e, 1);
        // lane lg holds the full 16-lane sum for row r = lg>>1 (lane pairs duplicate)
        if ((lane & 1) == 0) {
            orow_ws[rowbase + 8*lh + (lg >> 1)] = e + b3m;
        }
    }
}

// out[b][c] = sum_m o_ws[m][b] * Wout[c][m] + bout[c]
// One thread per batch row: o_ws read once, coalesced; Wout is wave-uniform (scalar loads).
__global__ __launch_bounds__(256) void nam_out_kernel(
    const float* __restrict__ o_ws, const float* __restrict__ Wout,
    const float* __restrict__ bout, float* __restrict__ out)
{
    const int b = blockIdx.x*256 + threadIdx.x;
    if (b >= BB) return;
    float acc[CLSC];
    #pragma unroll
    for (int c = 0; c < CLSC; ++c) acc[c] = bout[c];
    for (int mi = 0; mi < MM; ++mi) {
        const float val = o_ws[mi*BB + b];
        #pragma unroll
        for (int c = 0; c < CLSC; ++c)
            acc[c] = fmaf(val, Wout[c*MM + mi], acc[c]);
    }
    #pragma unroll
    for (int c = 0; c < CLSC; ++c) out[b*CLSC + c] = acc[c];
}

extern "C" void kernel_launch(void* const* d_in, const int* in_sizes, int n_in,
                              void* d_out, int out_size, void* d_ws, size_t ws_size,
                              hipStream_t stream) {
    const float* x        = (const float*)d_in[0];
    const int*   pair_idx = (const int*)  d_in[1];
    const float* W1   = (const float*)d_in[2];
    const float* b1   = (const float*)d_in[3];
    const float* g1   = (const float*)d_in[4];
    const float* be1  = (const float*)d_in[5];
    const float* m1   = (const float*)d_in[6];
    const float* v1   = (const float*)d_in[7];
    const float* W2   = (const float*)d_in[8];
    const float* b2   = (const float*)d_in[9];
    const float* g2   = (const float*)d_in[10];
    const float* be2  = (const float*)d_in[11];
    const float* m2   = (const float*)d_in[12];
    const float* v2   = (const float*)d_in[13];
    const float* W3   = (const float*)d_in[14];
    const float* b3   = (const float*)d_in[15];
    const float* Wout = (const float*)d_in[16];
    const float* bout = (const float*)d_in[17];
    float* out  = (float*)d_out;
    float* o_ws = (float*)d_ws;      // MM*BB floats (~4.06 MB), [m][b] layout

    dim3 grid1(MM, BB / BTILE);
    nam_pair_kernel<<<grid1, 256, 0, stream>>>(
        x, pair_idx, W1, b1, g1, be1, m1, v1,
        W2, b2, g2, be2, m2, v2, W3, b3, o_ws);

    nam_out_kernel<<<(BB + 255) / 256, 256, 0, stream>>>(o_ws, Wout, bout, out);
}
